// TransformerLayer_3848290697245
// MI455X (gfx1250) — matrix operand order repaired and checked by reading, not compiled, not run on hardware
//
#include <hip/hip_runtime.h>
#include <hip/hip_bf16.h>
#include <math.h>

#define DEPTH 2
#define DIM   512
#define HEADS 8
#define DH    64
#define INNER 512           // HEADS*DH
#define MLP   2048
#define BB    2
#define NN    2048
#define BN    4096          // BB*NN
#define EPS   1e-6f
#define NEGF  (-3.402823466e+38f)
#define SCALE 0.125f        // DH^-0.5

typedef __attribute__((ext_vector_type(16))) __bf16 bf16x16;
typedef __attribute__((ext_vector_type(8)))  float  f32x8;
typedef int v4i __attribute__((vector_size(16)));

union FragU { uint4 q[2]; bf16x16 v; };

__device__ inline f32x8 zero8() {
    f32x8 z;
#pragma unroll
    for (int e = 0; e < 8; ++e) z[e] = 0.f;
    return z;
}

// ---- CDNA5 async global->LDS copy of 16 bytes (per-lane), ASYNCcnt-tracked.
// Builtin signature (from compiler diagnostic): param0 = v4i32 in AS(1) (global),
// param1 = v4i32 in AS(3) (LDS), then imm offset, imm cpol.
__device__ inline void async_cp16(__bf16* dst_lds, const __bf16* src_glb) {
#if __has_builtin(__builtin_amdgcn_global_load_async_to_lds_b128)
    typedef __attribute__((address_space(1))) v4i GV;
    typedef __attribute__((address_space(3))) v4i LV;
    __builtin_amdgcn_global_load_async_to_lds_b128(
        (GV*)(unsigned long long)(uintptr_t)src_glb,
        (LV*)(unsigned)(uintptr_t)dst_lds,        // flat addr[31:0] == LDS offset
        0, 0);
#else
    unsigned           l = (unsigned)(uintptr_t)dst_lds;
    unsigned long long g = (unsigned long long)(uintptr_t)src_glb;
    asm volatile("global_load_async_to_lds_b128 %0, %1, off" :: "v"(l), "v"(g) : "memory");
#endif
}

__device__ inline void wait_async0() {
#if __has_builtin(__builtin_amdgcn_s_wait_asynccnt)
    __builtin_amdgcn_s_wait_asynccnt(0);
#else
    asm volatile("s_wait_asynccnt 0x0" ::: "memory");
#endif
}

// A fragment (16x32 bf16, M x K), LDS row-major [m][k], pitch in bf16 units.
__device__ inline bf16x16 frag_a(const __bf16* base, int pitch, int kt) {
    int lane = threadIdx.x & 31;
    int r = lane & 15, h = lane >> 4;
    const char* p = (const char*)(base + (size_t)r * pitch + kt + h * 8);
    FragU f;
    f.q[0] = *(const uint4*)p;
    f.q[1] = *(const uint4*)(p + 32);   // +16 bf16 => +32 bytes
    return f.v;
}

// B fragment (32x16 bf16, K x N), LDS stored transposed [n][k], pitch in bf16.
__device__ inline bf16x16 frag_b(const __bf16* base, int pitch, int kt) {
    int lane = threadIdx.x & 31;
    int c = lane & 15, h = lane >> 4;
    const char* p = (const char*)(base + (size_t)c * pitch + kt + h * 8);
    FragU f;
    f.q[0] = *(const uint4*)p;
    f.q[1] = *(const uint4*)(p + 32);
    return f.v;
}

__device__ inline f32x8 wmma_bf(bf16x16 a, bf16x16 b, f32x8 c) {
    return __builtin_amdgcn_wmma_f32_16x16x32_bf16(false, a, false, b, (short)0, c, false, false);
}

// ---------------------------------------------------------------- colstats
__global__ __launch_bounds__(256) void tl_colstats(const float* __restrict__ x,
                                                   float* __restrict__ mu,
                                                   float* __restrict__ var) {
    int col = blockIdx.x;                 // 0 .. BB*DIM-1
    int b = col / DIM, d = col % DIM;
    const float* p = x + (size_t)b * NN * DIM + d;
    float s = 0.f, ss = 0.f;
    for (int n = threadIdx.x; n < NN; n += 256) {
        float v = p[(size_t)n * DIM];
        s += v; ss += v * v;
    }
    __shared__ float sm[256], sq[256];
    sm[threadIdx.x] = s; sq[threadIdx.x] = ss;
    __syncthreads();
    for (int st = 128; st > 0; st >>= 1) {
        if (threadIdx.x < st) {
            sm[threadIdx.x] += sm[threadIdx.x + st];
            sq[threadIdx.x] += sq[threadIdx.x + st];
        }
        __syncthreads();
    }
    if (threadIdx.x == 0) {
        float m = sm[0] * (1.f / NN);
        mu[col]  = m;
        var[col] = sq[0] * (1.f / NN) - m * m;   // biased, as in batch_norm
    }
}

// ---------------------------------------------------------------- adain -> bf16
__global__ __launch_bounds__(256) void tl_adain_bf16(const float* __restrict__ x,
                                                     const float* __restrict__ mu,
                                                     const float* __restrict__ var,
                                                     const float* __restrict__ w,
                                                     const float* __restrict__ bp,
                                                     __bf16* __restrict__ out) {
    size_t i = (size_t)blockIdx.x * 256 + threadIdx.x;   // < BN*DIM
    int d = (int)(i % DIM);
    int b = (int)(i / ((size_t)NN * DIM));
    int col = b * DIM + d;
    float xn = (x[i] - mu[col]) * rsqrtf(var[col] + EPS);
    out[i] = (__bf16)(xn * w[col] + bp[col]);
}

// ---------------------------------------------------------------- WMMA GEMM
// C[M,Nn] = A[M,K](bf16) * B[K,Nn](f32 weights, converted on stage)
// EPI: 0 = f32 ; 1 = +bias +residual f32 ; 2 = +bias, GELU(erf) -> bf16 ; 3 = plain -> bf16
// Double-buffered LDS; A tile staged with async global->LDS b128 copies.
template <int EPI>
__global__ __launch_bounds__(256) void tl_gemm(const __bf16* __restrict__ A,
                                               const float*  __restrict__ Bw,
                                               const float*  __restrict__ bias,
                                               const float*  __restrict__ res,
                                               float*  __restrict__ Cf,
                                               __bf16* __restrict__ Cb,
                                               int M, int Nn, int K) {
    __shared__ __bf16 lA[2][128][40];   // [m][k], 8-short pad
    __shared__ __bf16 lB[2][128][40];   // [n][k]
    int tid  = threadIdx.x;
    int lane = tid & 31, wave = tid >> 5;
    int wm = wave >> 2, wn = wave & 3;           // 2 x 4 wave grid
    int bM = blockIdx.y * 128, bN = blockIdx.x * 128;

    f32x8 acc[4][2];
#pragma unroll
    for (int i = 0; i < 4; ++i)
#pragma unroll
        for (int j = 0; j < 2; ++j) acc[i][j] = zero8();

    auto stage = [&](int buf, int k0) {
        {   // A tile 128x32 bf16: pure copy -> async DMA into LDS
            int c = tid & 3, r0 = tid >> 2;                    // r0: 0..63
            async_cp16(&lA[buf][r0][c * 8],
                       A + (size_t)(bM + r0) * K + k0 + c * 8);
            async_cp16(&lA[buf][r0 + 64][c * 8],
                       A + (size_t)(bM + r0 + 64) * K + k0 + c * 8);
        }
        {   // B tile 32x128 f32 -> bf16, transposed into [n][k]
            int n4 = (tid & 31) * 4, kr = tid >> 5;            // kr: 0..7
#pragma unroll
            for (int i = 0; i < 4; ++i) {
                int k = kr + 8 * i;
                float4 v = *(const float4*)(Bw + (size_t)(k0 + k) * Nn + bN + n4);
                lB[buf][n4 + 0][k] = (__bf16)v.x; lB[buf][n4 + 1][k] = (__bf16)v.y;
                lB[buf][n4 + 2][k] = (__bf16)v.z; lB[buf][n4 + 3][k] = (__bf16)v.w;
            }
        }
    };

    stage(0, 0);
    wait_async0();
    __syncthreads();

    int nk = K >> 5;
    for (int t = 0; t < nk; ++t) {
        int cur = t & 1;
        if (t + 1 < nk) stage(cur ^ 1, (t + 1) << 5);   // prefetch next tile over compute

        bf16x16 af[4], bfr[2];
#pragma unroll
        for (int i = 0; i < 4; ++i) af[i]  = frag_a(&lA[cur][wm * 64 + i * 16][0], 40, 0);
#pragma unroll
        for (int j = 0; j < 2; ++j) bfr[j] = frag_b(&lB[cur][wn * 32 + j * 16][0], 40, 0);
#pragma unroll
        for (int i = 0; i < 4; ++i)
#pragma unroll
            for (int j = 0; j < 2; ++j) acc[i][j] = wmma_bf(af[i], bfr[j], acc[i][j]);

        wait_async0();
        __syncthreads();
    }

    // Epilogue. D layout: row = e + 8*(lane>>4), col = lane&15 within 16x16 tile.
    int r = lane & 15, h = lane >> 4;
#pragma unroll
    for (int i = 0; i < 4; ++i)
#pragma unroll
        for (int j = 0; j < 2; ++j) {
            int col = bN + wn * 32 + j * 16 + r;
            float bv = (EPI == 1 || EPI == 2) ? bias[col] : 0.f;
#pragma unroll
            for (int e = 0; e < 8; ++e) {
                int row = bM + wm * 64 + i * 16 + e + 8 * h;
                size_t idx = (size_t)row * Nn + col;
                float v = acc[i][j][e] + bv;
                if (EPI == 2) v = 0.5f * v * (1.0f + erff(v * 0.70710678118654752f));
                if (EPI == 1) v += res[idx];
                if (EPI == 2 || EPI == 3) Cb[idx] = (__bf16)v;
                else                      Cf[idx] = v;
            }
        }
}

// ---------------------------------------------------------------- attention
// grid (B*H, N/128); 8 waves, each owns 16 query rows. Flash-style online softmax.
// qkv is bf16 [BN][1536]; Q/K tiles staged with async LDS copies, V transposed on stage.
__global__ __launch_bounds__(256) void tl_attention(const __bf16* __restrict__ qkv,
                                                    const int*    __restrict__ mask,
                                                    __bf16* __restrict__ outbf) {
    __shared__ __bf16 lQ[128][72];
    __shared__ __bf16 lK[64][72];
    __shared__ __bf16 lVt[64][72];       // [dh][key] == B-fragment layout for PV
    __shared__ __bf16 lP[8][16][72];     // per-wave P scratch (A layout reload)
    __shared__ float  lMq[128], lMk[64];

    int bh = blockIdx.x;                 // 0..15
    int b = bh >> 3, hh = bh & 7;
    int q0 = blockIdx.y * 128;
    int tid = threadIdx.x, wave = tid >> 5, lane = tid & 31;
    int r = lane & 15, h = lane >> 4;

    const __bf16* qb = qkv + (size_t)b * NN * (3 * INNER) + hh * DH;
    const __bf16* kb = qb + INNER;
    const __bf16* vb = qb + 2 * INNER;

    for (int s = tid; s < 128 * 8; s += 256) {           // 16B chunks of Q rows
        int row = s >> 3, c = s & 7;
        async_cp16(&lQ[row][c * 8], qb + (size_t)(q0 + row) * (3 * INNER) + c * 8);
    }
    if (tid < 128) lMq[tid] = (float)mask[b * NN + q0 + tid];
    wait_async0();
    __syncthreads();

    bf16x16 aq0 = frag_a(&lQ[wave * 16][0], 72, 0);
    bf16x16 aq1 = frag_a(&lQ[wave * 16][0], 72, 32);
    float qm[8];
#pragma unroll
    for (int e = 0; e < 8; ++e) qm[e] = lMq[wave * 16 + e + 8 * h];

    float m_run[8], l_run[8];
    f32x8 o[4];
#pragma unroll
    for (int e = 0; e < 8; ++e) { m_run[e] = NEGF; l_run[e] = 0.f; }
#pragma unroll
    for (int n = 0; n < 4; ++n) o[n] = zero8();

    for (int kc = 0; kc < NN; kc += 64) {
        __syncthreads();
        for (int s = tid; s < 64 * 8; s += 256) {        // K rows: async copy
            int row = s >> 3, c = s & 7;
            async_cp16(&lK[row][c * 8], kb + (size_t)(kc + row) * (3 * INNER) + c * 8);
        }
        for (int s = tid; s < 64 * 8; s += 256) {        // V rows: load 8 bf16, transpose-store
            int row = s >> 3, c = s & 7;
            uint4 v4 = *(const uint4*)(vb + (size_t)(kc + row) * (3 * INNER) + c * 8);
            const __bf16* e8 = (const __bf16*)&v4;
#pragma unroll
            for (int j = 0; j < 8; ++j) lVt[c * 8 + j][row] = e8[j];
        }
        if (tid < 64) lMk[tid] = (float)mask[b * NN + kc + tid];
        wait_async0();
        __syncthreads();

        // S = Q(16x64) * K^T : 4 key-tiles x 2 k-steps
        f32x8 s[4];
#pragma unroll
        for (int j = 0; j < 4; ++j) {
            s[j] = wmma_bf(aq0, frag_b(&lK[j * 16][0], 72, 0),  zero8());
            s[j] = wmma_bf(aq1, frag_b(&lK[j * 16][0], 72, 32), s[j]);
        }
        // scale + mask
#pragma unroll
        for (int j = 0; j < 4; ++j) {
            float km = lMk[j * 16 + r];
#pragma unroll
            for (int e = 0; e < 8; ++e)
                s[j][e] = (qm[e] * km == 0.f) ? NEGF : s[j][e] * SCALE;
        }
        // online softmax: row max
        float cm[8];
#pragma unroll
        for (int e = 0; e < 8; ++e) {
            float v = fmaxf(fmaxf(s[0][e], s[1][e]), fmaxf(s[2][e], s[3][e]));
            v = fmaxf(v, __shfl_xor(v, 1));
            v = fmaxf(v, __shfl_xor(v, 2));
            v = fmaxf(v, __shfl_xor(v, 4));
            v = fmaxf(v, __shfl_xor(v, 8));
            cm[e] = v;
        }
        float alpha[8], lc[8];
#pragma unroll
        for (int e = 0; e < 8; ++e) {
            float mn = fmaxf(m_run[e], cm[e]);
            alpha[e] = __expf(m_run[e] - mn);
            m_run[e] = mn;
            lc[e] = 0.f;
        }
        // p = exp(s - m), store into per-wave LDS in A layout [row][key]
#pragma unroll
        for (int j = 0; j < 4; ++j)
#pragma unroll
            for (int e = 0; e < 8; ++e) {
                float p = __expf(s[j][e] - m_run[e]);
                lc[e] += p;
                lP[wave][e + 8 * h][j * 16 + r] = (__bf16)p;
            }
#pragma unroll
        for (int e = 0; e < 8; ++e) {
            float v = lc[e];
            v += __shfl_xor(v, 1);
            v += __shfl_xor(v, 2);
            v += __shfl_xor(v, 4);
            v += __shfl_xor(v, 8);
            l_run[e] = l_run[e] * alpha[e] + v;
        }
#pragma unroll
        for (int n = 0; n < 4; ++n)
#pragma unroll
            for (int e = 0; e < 8; ++e) o[n][e] *= alpha[e];

        __asm__ volatile("" ::: "memory");   // keep lP stores before reload (DS in-order per wave)

        bf16x16 ap0 = frag_a(&lP[wave][0][0], 72, 0);
        bf16x16 ap1 = frag_a(&lP[wave][0][0], 72, 32);
#pragma unroll
        for (int n = 0; n < 4; ++n) {
            o[n] = wmma_bf(ap0, frag_b(&lVt[n * 16][0], 72, 0),  o[n]);
            o[n] = wmma_bf(ap1, frag_b(&lVt[n * 16][0], 72, 32), o[n]);
        }
    }

#pragma unroll
    for (int n = 0; n < 4; ++n)
#pragma unroll
        for (int e = 0; e < 8; ++e) {
            int row = q0 + wave * 16 + e + 8 * h;
            int col = hh * DH + n * 16 + r;
            outbf[(size_t)(b * NN + row) * INNER + col] = (__bf16)(o[n][e] / l_run[e]);
        }
}

// ---------------------------------------------------------------- launcher
extern "C" void kernel_launch(void* const* d_in, const int* in_sizes, int n_in,
                              void* d_out, int out_size, void* d_ws, size_t ws_size,
                              hipStream_t stream) {
    const float* x_in  = (const float*)d_in[0];
    const int*   mask  = (const int*)  d_in[1];
    const float* qkv_w = (const float*)d_in[2];
    const float* out_w = (const float*)d_in[3];
    const float* out_b = (const float*)d_in[4];
    const float* ff_w1 = (const float*)d_in[5];
    const float* ff_b1 = (const float*)d_in[6];
    const float* ff_w2 = (const float*)d_in[7];
    const float* ff_b2 = (const float*)d_in[8];
    const float* ad1_w = (const float*)d_in[9];
    const float* ad1_b = (const float*)d_in[10];
    const float* ad2_w = (const float*)d_in[11];
    const float* ad2_b = (const float*)d_in[12];
    float* outp = (float*)d_out;

    // workspace carve-out (~53 MB)
    char* p = (char*)d_ws;
    auto take = [&](size_t bytes) {
        char* r = p;
        p += (bytes + 255) & ~(size_t)255;
        return r;
    };
    float*  mu     = (float*) take((size_t)BB * DIM * 4);
    float*  var_   = (float*) take((size_t)BB * DIM * 4);
    __bf16* hbf    = (__bf16*)take((size_t)BN * DIM * 2);
    __bf16* qkvbf  = (__bf16*)take((size_t)BN * 3 * INNER * 2);
    __bf16* attnbf = (__bf16*)take((size_t)BN * INNER * 2);
    __bf16* mlpbf  = (__bf16*)take((size_t)BN * MLP * 2);
    float*  xmid   = (float*) take((size_t)BN * DIM * 4);
    float*  x1     = (float*) take((size_t)BN * DIM * 4);

    const int EW = 256;
    dim3 blk(EW);

    for (int l = 0; l < DEPTH; ++l) {
        const float* xc    = (l == 0) ? x_in : x1;
        float*       xnext = (l == DEPTH - 1) ? outp : x1;

        // --- AdaIN 1 -> bf16 activation
        tl_colstats<<<BB * DIM, blk, 0, stream>>>(xc, mu, var_);
        tl_adain_bf16<<<(BN * DIM) / EW, blk, 0, stream>>>(
            xc, mu, var_, ad1_w + (size_t)l * BB * DIM, ad1_b + (size_t)l * BB * DIM, hbf);

        // --- QKV projection: [4096,512] x [512,1536] -> bf16 (feeds WMMA attention)
        tl_gemm<3><<<dim3(3 * INNER / 128, BN / 128), blk, 0, stream>>>(
            hbf, qkv_w + (size_t)l * DIM * 3 * INNER, nullptr, nullptr,
            nullptr, qkvbf, BN, 3 * INNER, DIM);

        // --- attention (flash, masked) -> bf16
        tl_attention<<<dim3(BB * HEADS, NN / 128), blk, 0, stream>>>(qkvbf, mask, attnbf);

        // --- output projection + bias + residual -> xmid (f32)
        tl_gemm<1><<<dim3(DIM / 128, BN / 128), blk, 0, stream>>>(
            attnbf, out_w + (size_t)l * INNER * DIM, out_b + (size_t)l * DIM, xc,
            xmid, nullptr, BN, DIM, INNER);

        // --- AdaIN 2 -> bf16
        tl_colstats<<<BB * DIM, blk, 0, stream>>>(xmid, mu, var_);
        tl_adain_bf16<<<(BN * DIM) / EW, blk, 0, stream>>>(
            xmid, mu, var_, ad2_w + (size_t)l * BB * DIM, ad2_b + (size_t)l * BB * DIM, hbf);

        // --- FF1 + bias + exact GELU -> bf16
        tl_gemm<2><<<dim3(MLP / 128, BN / 128), blk, 0, stream>>>(
            hbf, ff_w1 + (size_t)l * DIM * MLP, ff_b1 + (size_t)l * MLP, nullptr,
            nullptr, mlpbf, BN, MLP, DIM);

        // --- FF2 + bias + residual -> xnext (f32)
        tl_gemm<1><<<dim3(DIM / 128, BN / 128), blk, 0, stream>>>(
            mlpbf, ff_w2 + (size_t)l * MLP * DIM, ff_b2 + (size_t)l * DIM, xmid,
            xnext, nullptr, BN, DIM, MLP);
    }
}
